// multihead_self_attention_40561671143945
// MI455X (gfx1250) — compile-verified, matrix operand order checked by reading, not run on hardware
//
#include <hip/hip_runtime.h>

typedef unsigned short u16;
typedef unsigned int   u32;
typedef unsigned long long u64;
typedef __attribute__((ext_vector_type(16))) __bf16 v16bf;
typedef __attribute__((ext_vector_type(8)))  float  v8f;
typedef __attribute__((ext_vector_type(4)))  unsigned int ui32x4;
typedef __attribute__((ext_vector_type(8)))  int i32x8;
typedef __attribute__((ext_vector_type(4)))  int i32x4;

#define D_MODEL   1024
#define NUM_HEADS 16
#define SEQ       2048
#define BATCH     2
#define DHEAD     64

struct U32x8 { uint4 a, b; };

__device__ __forceinline__ u16 f2bf(float f) {
  // round-to-nearest-even bf16 truncation
  u32 u = __float_as_uint(f);
  u32 r = u + 0x7FFFu + ((u >> 16) & 1u);
  return (u16)(r >> 16);
}

__device__ __forceinline__ v16bf load_frag(const u16* base, int kb0) {
  U32x8 t;
  t.a = *(const uint4*)(base + kb0);        // K = kb0 .. kb0+7
  t.b = *(const uint4*)(base + kb0 + 16);   // K = kb0+16 .. kb0+23
  return __builtin_bit_cast(v16bf, t);
}

__device__ __forceinline__ v8f v8f_zero() {
  v8f z = {0.f, 0.f, 0.f, 0.f, 0.f, 0.f, 0.f, 0.f};
  return z;
}

// ------------------------------------------------------------------ TDM
// Issue a 2-D tile DMA global->LDS via the Tensor Data Mover.
// tile_w elements per row (bf16, data_size=2B), tile_h rows,
// row_stride = tensor row pitch in elements. Wave-uniform args required.
__device__ __forceinline__ void tdm_load_2d(u32 lds_off, const void* gptr,
                                            u32 tile_w, u32 tile_h,
                                            u32 row_stride) {
  u64 ga = (u64)(uintptr_t)gptr;
  ui32x4 g0;
  g0.x = 1u;                                            // count=1, user mode
  g0.y = lds_off;                                       // D#.lds_addr (bytes)
  g0.z = (u32)(ga & 0xFFFFFFFFu);                       // global_addr[31:0]
  g0.w = (u32)((ga >> 32) & 0x01FFFFFFu) | (2u << 30);  // addr[56:32] | type=2
  i32x8 g1;
  g1[0] = (int)(1u << 16);                              // data_size=1 (2 bytes)
  g1[1] = (int)((tile_w & 0xFFFFu) << 16);              // tensor_dim0[15:0]
  g1[2] = (int)((tile_w >> 16) | ((tile_h & 0xFFFFu) << 16)); // dim0[31:16]|dim1[15:0]
  g1[3] = (int)((tile_h >> 16) | (tile_w << 16));       // dim1[31:16] | tile_dim0
  g1[4] = (int)(tile_h & 0xFFFFu);                      // tile_dim1 (tile_dim2=0)
  g1[5] = (int)row_stride;                              // tensor_dim0_stride[31:0]
  g1[6] = 0;                                            // stride[47:32]|dim1_stride lo
  g1[7] = 0;
  i32x4 z4 = {0, 0, 0, 0};
#if __clang_major__ >= 23
  i32x8 z8 = {0, 0, 0, 0, 0, 0, 0, 0};
  __builtin_amdgcn_tensor_load_to_lds(g0, g1, z4, z4, z8, 0);
#else
  __builtin_amdgcn_tensor_load_to_lds(g0, g1, z4, z4, 0);
#endif
}

// ---------------------------------------------------------------- convert
__global__ __launch_bounds__(256) void cvt_f32_bf16(const float* __restrict__ src,
                                                    u16* __restrict__ dst, int n) {
  int i = blockIdx.x * 256 + threadIdx.x;
  if (i < n) dst[i] = f2bf(src[i]);
}

// ---------------------------------------------------------------- GEMM
// C[M,N](f32) = A[M,K](bf16) * W[N,K](bf16)^T ; M%128==0, N%128==0, K%32==0
// Block tile 128x128, 8 waves, wave tile 32x64 (8 WMMA / K-step).
// TDM double-buffered staging: wave 0 DMAs the next K-slice while all waves
// run WMMAs on the current one.
__global__ __launch_bounds__(256) void gemm_bf16(const u16* __restrict__ A,
                                                 const u16* __restrict__ Bw,
                                                 float* __restrict__ C,
                                                 int M, int N, int K) {
  __shared__ alignas(16) u16 lA[2][128 * 32];
  __shared__ alignas(16) u16 lB[2][128 * 32];
  const int tid  = threadIdx.x;
  const int wid  = tid >> 5;
  const int lane = tid & 31;
  const int ln15 = lane & 15;
  const int hiv  = lane >> 4;
  const int kb0  = hiv << 3;          // 0 or 8 (K sub-block per lane half)
  const int wm   = wid & 3;           // 4 wave-rows of 32
  const int wn   = wid >> 2;          // 2 wave-cols of 64
  const int m0   = blockIdx.x * 128;
  const int n0   = blockIdx.y * 128;
  const int iters = K >> 5;

  v8f c[2][4];
#pragma unroll
  for (int hf = 0; hf < 2; ++hf)
#pragma unroll
    for (int j = 0; j < 4; ++j) c[hf][j] = v8f_zero();

  if (wid == 0) {  // preload slice 0 into buffer 0
    tdm_load_2d((u32)(uintptr_t)&lA[0][0], A + (size_t)m0 * K, 32, 128, (u32)K);
    tdm_load_2d((u32)(uintptr_t)&lB[0][0], Bw + (size_t)n0 * K, 32, 128, (u32)K);
  }

  for (int it = 0; it < iters; ++it) {
    const int cur = it & 1;
    if (wid == 0) {
      if (it + 1 < iters) {
        const int k0 = (it + 1) << 5;
        tdm_load_2d((u32)(uintptr_t)&lA[cur ^ 1][0],
                    A + (size_t)m0 * K + k0, 32, 128, (u32)K);
        tdm_load_2d((u32)(uintptr_t)&lB[cur ^ 1][0],
                    Bw + (size_t)n0 * K + k0, 32, 128, (u32)K);
        __builtin_amdgcn_s_wait_tensorcnt(2);  // cur's 2 DMAs done, next in flight
      } else {
        __builtin_amdgcn_s_wait_tensorcnt(0);
      }
    }
    __syncthreads();   // cur buffer visible to all waves

    v16bf a0 = load_frag(&lA[cur][(wm * 32 + ln15) * 32], kb0);
    v16bf a1 = load_frag(&lA[cur][(wm * 32 + 16 + ln15) * 32], kb0);
#pragma unroll
    for (int j = 0; j < 4; ++j) {
      v16bf b = load_frag(&lB[cur][(wn * 64 + j * 16 + ln15) * 32], kb0);
      c[0][j] = __builtin_amdgcn_wmma_f32_16x16x32_bf16(false, a0, false, b,
                                                        (short)0, c[0][j], false, false);
      c[1][j] = __builtin_amdgcn_wmma_f32_16x16x32_bf16(false, a1, false, b,
                                                        (short)0, c[1][j], false, false);
    }
    __syncthreads();   // all reads of cur done before it is re-targeted
  }

#pragma unroll
  for (int hf = 0; hf < 2; ++hf) {
    const int rbase = m0 + wm * 32 + hf * 16 + (hiv << 3);
#pragma unroll
    for (int j = 0; j < 4; ++j) {
      int colg = n0 + wn * 64 + j * 16 + ln15;
#pragma unroll
      for (int r = 0; r < 8; ++r)
        C[(size_t)(rbase + r) * N + colg] = c[hf][j][r];
    }
  }
}

// ---------------------------------------------------------------- RoPE+pack
// qkv f32 [B,S,3D] -> Qb/Kb bf16 [B,H,S,64] (RoPE'd), Vt bf16 [B,H,64,S]
__global__ __launch_bounds__(256) void rope_pack(const float* __restrict__ qkv,
                                                 const int* __restrict__ pos,
                                                 u16* __restrict__ Qb,
                                                 u16* __restrict__ Kb,
                                                 u16* __restrict__ Vt) {
  int idx = blockIdx.x * 256 + threadIdx.x;      // 2^21 threads total
  int j = idx & 31;
  int s = (idx >> 5) & (SEQ - 1);
  int h = (idx >> 16) & (NUM_HEADS - 1);
  int b = idx >> 20;

  float p   = (float)pos[s];
  float inv = __expf(-(float)j * 0.2878231366242582f);  // ln(10000)/32
  float sn, cs;
  __sincosf(p * inv, &sn, &cs);

  size_t base = (size_t)(b * SEQ + s) * (3 * D_MODEL);
  int hd = h * DHEAD + 2 * j;
  float q0 = qkv[base + hd],               q1 = qkv[base + hd + 1];
  float k0 = qkv[base + D_MODEL + hd],     k1 = qkv[base + D_MODEL + hd + 1];
  float v0 = qkv[base + 2 * D_MODEL + hd], v1 = qkv[base + 2 * D_MODEL + hd + 1];

  size_t qi = (size_t)((b * NUM_HEADS + h) * SEQ + s) * DHEAD + 2 * j;
  Qb[qi]     = f2bf(q0 * cs - q1 * sn);
  Qb[qi + 1] = f2bf(q0 * sn + q1 * cs);
  Kb[qi]     = f2bf(k0 * cs - k1 * sn);
  Kb[qi + 1] = f2bf(k0 * sn + k1 * cs);

  size_t vb = (size_t)(b * NUM_HEADS + h) * DHEAD * SEQ;
  Vt[vb + (size_t)(2 * j) * SEQ + s]     = f2bf(v0);
  Vt[vb + (size_t)(2 * j + 1) * SEQ + s] = f2bf(v1);
}

// ---------------------------------------------------------------- flash attn
// Causal online-softmax attention; out attn bf16 [B,S,D].
// Fully per-wave pipeline: each wave DMAs its own 32-key K/V tiles into
// private LDS via TDM (per-wave TENSORcnt), overlapping the next block's DMA
// with the current block's WMMAs + softmax. No workgroup barriers at all.
__global__ __launch_bounds__(128) void flash_attn(const u16* __restrict__ Q,
                                                  const u16* __restrict__ K,
                                                  const u16* __restrict__ Vt,
                                                  u16* __restrict__ attn) {
  __shared__ alignas(16) u16 lk[4][32 * 64];   // per-wave K tile (keys x dh) 4KB
  __shared__ alignas(16) u16 lv[4][64 * 32];   // per-wave V tile (dh x keys) 4KB
  __shared__ alignas(16) u16 lp[4][16 * 32];   // per-wave P staging        1KB
  const int bh   = blockIdx.y;
  const int b    = bh >> 4, h = bh & 15;
  const int wid  = threadIdx.x >> 5;
  const int lane = threadIdx.x & 31;
  const int ln15 = lane & 15;
  const int hiv  = lane >> 4;
  const int kb0  = hiv << 3;
  const int q0   = blockIdx.x * 64 + wid * 16;

  const u16* Qp = Q  + (size_t)bh * SEQ * DHEAD;
  const u16* Kp = K  + (size_t)bh * SEQ * DHEAD;
  const u16* Vp = Vt + (size_t)bh * DHEAD * SEQ;

  const u16* qr = Qp + (size_t)(q0 + ln15) * DHEAD;
  v16bf qa0 = load_frag(qr, kb0);        // dh 0..31
  v16bf qa1 = load_frag(qr + 32, kb0);   // dh 32..63

  float mi[8], li[8];
  v8f acc[4];
#pragma unroll
  for (int r = 0; r < 8; ++r) { mi[r] = -3.0e38f; li[r] = 0.f; }
#pragma unroll
  for (int j = 0; j < 4; ++j) acc[j] = v8f_zero();

  u16* lkw = lk[wid];
  u16* lvw = lv[wid];
  u16* lpw = lp[wid];
  const int nkb = (q0 + 16 + 31) >> 5;   // 32-key blocks needed (causal)

  // prologue: DMA block 0 K/V tiles (per-wave TDM)
  tdm_load_2d((u32)(uintptr_t)lkw, Kp, 64, 32, DHEAD);
  tdm_load_2d((u32)(uintptr_t)lvw, Vp, 32, 64, SEQ);

  for (int kb = 0; kb < nkb; ++kb) {
    const int kstart = kb << 5;
    __builtin_amdgcn_s_wait_tensorcnt(0);    // this wave's K/V tiles landed

    // pull all K/V fragments for this block into registers
    v16bf kf[2][2];
#pragma unroll
    for (int t = 0; t < 2; ++t) {
      kf[t][0] = load_frag(&lkw[(t * 16 + ln15) * 64], kb0);
      kf[t][1] = load_frag(&lkw[(t * 16 + ln15) * 64 + 32], kb0);
    }
    v16bf vf[4];
#pragma unroll
    for (int j = 0; j < 4; ++j)
      vf[j] = load_frag(&lvw[(j * 16 + ln15) * 32], kb0);

    // LDS reads complete -> the tiles are reusable; start next block's DMA
    asm volatile("s_wait_dscnt 0x0" ::: "memory");
    if (kb + 1 < nkb) {
      const int nk = (kb + 1) << 5;
      tdm_load_2d((u32)(uintptr_t)lkw, Kp + (size_t)nk * DHEAD, 64, 32, DHEAD);
      tdm_load_2d((u32)(uintptr_t)lvw, Vp + nk, 32, 64, SEQ);
    }

    // scores for two 16-key tiles
    v8f s01[2];
#pragma unroll
    for (int t = 0; t < 2; ++t) {
      const int kt = kstart + t * 16;
      v8f sc = v8f_zero();
      sc = __builtin_amdgcn_wmma_f32_16x16x32_bf16(false, qa0, false, kf[t][0],
                                                   (short)0, sc, false, false);
      sc = __builtin_amdgcn_wmma_f32_16x16x32_bf16(false, qa1, false, kf[t][1],
                                                   (short)0, sc, false, false);
      const int cg = kt + ln15;
#pragma unroll
      for (int r = 0; r < 8; ++r) {
        const int rg = q0 + r + (hiv << 3);
        sc[r] = (cg <= rg) ? sc[r] * 0.125f : -3.0e38f;
      }
      s01[t] = sc;
    }

    // online softmax (reductions across the 16-lane row groups)
    float sscale[8];
#pragma unroll
    for (int r = 0; r < 8; ++r) {
      float m = fmaxf(s01[0][r], s01[1][r]);
#pragma unroll
      for (int msk = 8; msk >= 1; msk >>= 1)
        m = fmaxf(m, __shfl_xor(m, msk, 16));
      float mnew = fmaxf(mi[r], m);
      sscale[r] = __expf(mi[r] - mnew);
      float p0 = __expf(s01[0][r] - mnew);
      float p1 = __expf(s01[1][r] - mnew);
      s01[0][r] = p0; s01[1][r] = p1;
      float rs = p0 + p1;
#pragma unroll
      for (int msk = 8; msk >= 1; msk >>= 1)
        rs += __shfl_xor(rs, msk, 16);
      li[r] = li[r] * sscale[r] + rs;
      mi[r] = mnew;
    }
#pragma unroll
    for (int j = 0; j < 4; ++j)
#pragma unroll
      for (int r = 0; r < 8; ++r) acc[j][r] *= sscale[r];

    // C-layout -> A-layout transpose of P via per-wave LDS (DS ops in-order)
#pragma unroll
    for (int r = 0; r < 8; ++r) {
      const int pr = r + (hiv << 3);
      lpw[pr * 32 + ln15]      = f2bf(s01[0][r]);
      lpw[pr * 32 + 16 + ln15] = f2bf(s01[1][r]);
    }
    v16bf pf = load_frag(&lpw[ln15 * 32], kb0);

#pragma unroll
    for (int j = 0; j < 4; ++j)
      acc[j] = __builtin_amdgcn_wmma_f32_16x16x32_bf16(false, pf, false, vf[j],
                                                       (short)0, acc[j], false, false);
  }

  const int rbase = q0 + (hiv << 3);
#pragma unroll
  for (int r = 0; r < 8; ++r) {
    float invl = 1.0f / li[r];
#pragma unroll
    for (int j = 0; j < 4; ++j)
      attn[(size_t)(b * SEQ + rbase + r) * D_MODEL + h * DHEAD + j * 16 + ln15] =
          f2bf(acc[j][r] * invl);
  }
}

// ---------------------------------------------------------------- launch
extern "C" void kernel_launch(void* const* d_in, const int* in_sizes, int n_in,
                              void* d_out, int out_size, void* d_ws, size_t ws_size,
                              hipStream_t stream) {
  (void)in_sizes; (void)n_in; (void)out_size; (void)ws_size;
  const float* X    = (const float*)d_in[0];
  const int*   tpos = (const int*)d_in[1];
  const float* Wqkv = (const float*)d_in[2];
  const float* Wo   = (const float*)d_in[3];
  float* out = (float*)d_out;

  const size_t nX  = (size_t)BATCH * SEQ * D_MODEL;   // 4,194,304
  const size_t nWq = (size_t)3 * D_MODEL * D_MODEL;   // 3,145,728
  const size_t nWo = (size_t)D_MODEL * D_MODEL;       // 1,048,576

  char* ws = (char*)d_ws;
  u16*   Xbf  = (u16*)ws;   ws += nX * 2;
  u16*   Wqbf = (u16*)ws;   ws += nWq * 2;
  u16*   Wobf = (u16*)ws;   ws += nWo * 2;
  float* QKV  = (float*)ws; ws += nX * 3 * 4;
  u16*   Qb   = (u16*)ws;   ws += nX * 2;
  u16*   Kb   = (u16*)ws;   ws += nX * 2;
  u16*   Vt   = (u16*)ws;   ws += nX * 2;
  u16*   Attn = (u16*)ws;   ws += nX * 2;           // ~101 MB total

  cvt_f32_bf16<<<(int)((nX  + 255) / 256), 256, 0, stream>>>(X,    Xbf,  (int)nX);
  cvt_f32_bf16<<<(int)((nWq + 255) / 256), 256, 0, stream>>>(Wqkv, Wqbf, (int)nWq);
  cvt_f32_bf16<<<(int)((nWo + 255) / 256), 256, 0, stream>>>(Wo,   Wobf, (int)nWo);

  gemm_bf16<<<dim3(BATCH * SEQ / 128, 3 * D_MODEL / 128), 256, 0, stream>>>(
      Xbf, Wqbf, QKV, BATCH * SEQ, 3 * D_MODEL, D_MODEL);

  rope_pack<<<(BATCH * NUM_HEADS * SEQ * 32) / 256, 256, 0, stream>>>(
      QKV, tpos, Qb, Kb, Vt);

  flash_attn<<<dim3(SEQ / 64, BATCH * NUM_HEADS), 128, 0, stream>>>(Qb, Kb, Vt, Attn);

  gemm_bf16<<<dim3(BATCH * SEQ / 128, D_MODEL / 128), 256, 0, stream>>>(
      Attn, Wobf, out, BATCH * SEQ, D_MODEL, D_MODEL);
}
